// PA_MSA_39152921870569
// MI455X (gfx1250) — matrix operand order repaired and checked by reading, not compiled, not run on hardware
//
#include <hip/hip_runtime.h>

// ---------------------------------------------------------------------------
// Restormer channel-attention (MDTA) fused pipeline for MI455X (gfx1250).
// All GEMMs use v_wmma_f32_16x16x32_bf16 (wave32 WMMA). fp32 accumulate.
// ---------------------------------------------------------------------------

typedef __bf16 bf16_t;
typedef __attribute__((ext_vector_type(8)))  __bf16 v8bf;
typedef __attribute__((ext_vector_type(16))) __bf16 v16bf;
typedef __attribute__((ext_vector_type(8)))  float  v8f;

union V16U { v16bf v; v8bf h[2]; };

#define B_    4
#define C_    192
#define HD_   4
#define CH_   48        // channels per head
#define N_    16384     // h*w = 128*128
#define NPIX  65536     // B_*N_
#define KSPLIT 64

static __device__ inline v8f wmma_bf16(v16bf a, v16bf b, v8f c) {
  // 8 args: (neg_a, A, neg_b, B, c_mod, C, reuse_a, reuse_b)
  return __builtin_amdgcn_wmma_f32_16x16x32_bf16(false, a, false, b,
                                                 (short)0, c, false, false);
}

static __device__ inline v8f vzero8f() {
  v8f z = {0.f,0.f,0.f,0.f,0.f,0.f,0.f,0.f};
  return z;
}

// ---------------------------------------------------------------------------
// Stage 0a: x (b,192,n) f32  ->  xT (b*n, 192) bf16   (pixel-major transpose)
// ---------------------------------------------------------------------------
__global__ __launch_bounds__(256)
void k_xt(const float* __restrict__ x, bf16_t* __restrict__ xT) {
  int g = blockIdx.x * 256 + threadIdx.x;     // global pixel 0..65535
  int b = g >> 14, p = g & 16383;
  const float* src = x + (size_t)b * C_ * N_ + p;
  bf16_t* dst = xT + (size_t)g * C_;
  for (int c0 = 0; c0 < C_; c0 += 8) {
    v8bf t;
#pragma unroll
    for (int i = 0; i < 8; ++i) t[i] = (bf16_t)src[(size_t)(c0 + i) * N_];
    *(v8bf*)(dst + c0) = t;
  }
}

// ---------------------------------------------------------------------------
// Stage 0b: concat+convert wq|wk|wv (192x192 f32 each) -> Wc (576x192 bf16)
// ---------------------------------------------------------------------------
__global__ __launch_bounds__(256)
void k_wc(const float* __restrict__ wq, const float* __restrict__ wk,
          const float* __restrict__ wv, bf16_t* __restrict__ Wc) {
  int i = blockIdx.x * 256 + threadIdx.x;     // < 576*192
  int r = i / (C_ * C_), rem = i % (C_ * C_);
  const float* s = (r == 0) ? wq : (r == 1) ? wk : wv;
  Wc[i] = (bf16_t)s[rem];
}

// ---------------------------------------------------------------------------
// Stage 1: qkv_raw(576 x 65536) = Wc(576x192) @ xT^T   [bf16 WMMA, K=192]
// Block 256 thr = 8 waves; block tile M=64,N=128; wave tile 16x64.
// ---------------------------------------------------------------------------
__global__ __launch_bounds__(256)
void k_gemm1(const bf16_t* __restrict__ Wc, const bf16_t* __restrict__ xT,
             bf16_t* __restrict__ raw) {
  int tid = threadIdx.x, lane = tid & 31, wave = tid >> 5;
  int sel = lane >> 4, lm = lane & 15;
  int m_t   = blockIdx.y * 64 + (wave >> 1) * 16;
  int nbase = blockIdx.x * 128 + (wave & 1) * 64;

  const bf16_t* arow = Wc + (size_t)(m_t + lm) * C_;
  const bf16_t* bcol[4];
#pragma unroll
  for (int t = 0; t < 4; ++t)
    bcol[t] = xT + (size_t)(nbase + t * 16 + lm) * C_;

  v8f acc[4];
#pragma unroll
  for (int t = 0; t < 4; ++t) acc[t] = vzero8f();

#pragma unroll
  for (int kk = 0; kk < 6; ++kk) {
    int k0 = kk * 32;
    V16U a;
    a.h[0] = *(const v8bf*)(arow + k0 + sel * 16);
    a.h[1] = *(const v8bf*)(arow + k0 + sel * 16 + 8);
    if (kk < 5) __builtin_prefetch(arow + k0 + 32, 0, 3);  // global_prefetch_b8
#pragma unroll
    for (int t = 0; t < 4; ++t) {
      V16U bb;
      bb.v = *(const v16bf*)(bcol[t] + k0 + sel * 16);
      acc[t] = wmma_bf16(a.v, bb.v, acc[t]);
    }
  }

#pragma unroll
  for (int t = 0; t < 4; ++t) {
    int j = nbase + t * 16 + lm;
    int b = j >> 14, p = j & 16383;
#pragma unroll
    for (int g = 0; g < 8; ++g) {
      int m = m_t + g + sel * 8;
      int r = m / C_, co = m % C_;
      raw[(((size_t)r * B_ + b) * C_ + co) * N_ + p] = (bf16_t)acc[t][g];
    }
  }
}

// ---------------------------------------------------------------------------
// Stage 2: depthwise 3x3 (zero pad) on qkv_raw -> qkv_dw (same layout, bf16)
// One thread per output pixel; block = 2 rows of one (r,b,c) plane.
// ---------------------------------------------------------------------------
__global__ __launch_bounds__(256)
void k_dw(const bf16_t* __restrict__ raw, const float* __restrict__ wq_dw,
          const float* __restrict__ wk_dw, const float* __restrict__ wv_dw,
          bf16_t* __restrict__ dwo) {
  int pr = blockIdx.x >> 6;                   // plane: r*768 + b*192 + c
  int rowpair = blockIdx.x & 63;
  int r = pr / (B_ * C_);
  int c = (pr % (B_ * C_)) % C_;
  int y = rowpair * 2 + (threadIdx.x >> 7);
  int xx0 = threadIdx.x & 127;
  const float* wsel = (r == 0) ? wq_dw : (r == 1) ? wk_dw : wv_dw;
  float w9[9];
#pragma unroll
  for (int i = 0; i < 9; ++i) w9[i] = wsel[c * 9 + i];
  const bf16_t* in = raw + (size_t)pr * N_;
  float acc = 0.f;
#pragma unroll
  for (int dy = -1; dy <= 1; ++dy) {
    int yy = y + dy;
    if (yy < 0 || yy > 127) continue;
#pragma unroll
    for (int dx = -1; dx <= 1; ++dx) {
      int xc = xx0 + dx;
      if (xc < 0 || xc > 127) continue;
      acc += w9[(dy + 1) * 3 + (dx + 1)] * (float)in[yy * 128 + xc];
    }
  }
  dwo[(size_t)pr * N_ + y * 128 + xx0] = (bf16_t)acc;
}

// ---------------------------------------------------------------------------
// Stage 3: inverse L2 norms of q rows (0..767) and k rows (768..1535)
// ---------------------------------------------------------------------------
__global__ __launch_bounds__(256)
void k_norm(const bf16_t* __restrict__ dwb, float* __restrict__ ninv) {
  __shared__ float red[256];
  int row = blockIdx.x;
  const bf16_t* src = dwb + (size_t)row * N_;
  float s = 0.f;
  for (int i = threadIdx.x; i < N_; i += 256) {
    float v = (float)src[i];
    s += v * v;
  }
  red[threadIdx.x] = s;
  __syncthreads();
  for (int st = 128; st > 0; st >>= 1) {
    if (threadIdx.x < st) red[threadIdx.x] += red[threadIdx.x + st];
    __syncthreads();
  }
  if (threadIdx.x == 0) ninv[row] = 1.f / fmaxf(sqrtf(red[0]), 1e-12f);
}

// ---------------------------------------------------------------------------
// Stage 4: split-K Gram  S_part[bh][split][48][48] = q_slice @ k_slice^T
// Block 288 thr = 9 waves, one 16x16 C tile each (3x3 tiles), K-slice = 256.
// ---------------------------------------------------------------------------
__global__ __launch_bounds__(288)
void k_gram(const bf16_t* __restrict__ dwb, float* __restrict__ Sp) {
  int tid = threadIdx.x, lane = tid & 31, wave = tid >> 5;  // wave 0..8
  int sel = lane >> 4, lm = lane & 15;
  int mt = wave / 3, nt = wave % 3;
  int bh = blockIdx.y, split = blockIdx.x;
  int b = bh >> 2, hd = bh & 3;
  const bf16_t* qrow = dwb + (size_t)(b * C_ + hd * CH_ + mt * 16 + lm) * N_;
  const bf16_t* krow =
      dwb + (size_t)(B_ * C_ + b * C_ + hd * CH_ + nt * 16 + lm) * N_;
  int pbase = split * 256;
  v8f acc = vzero8f();
#pragma unroll
  for (int kk = 0; kk < 8; ++kk) {
    int p0 = pbase + kk * 32;
    V16U a;
    a.h[0] = *(const v8bf*)(qrow + p0 + sel * 16);
    a.h[1] = *(const v8bf*)(qrow + p0 + sel * 16 + 8);
    V16U bb;
    bb.v = *(const v16bf*)(krow + p0 + sel * 16);
    acc = wmma_bf16(a.v, bb.v, acc);
  }
#pragma unroll
  for (int g = 0; g < 8; ++g) {
    int crow = mt * 16 + g + sel * 8;
    Sp[(((size_t)bh * KSPLIT + split) * CH_ + crow) * CH_ + nt * 16 + lm] =
        acc[g];
  }
}

// ---------------------------------------------------------------------------
// Stage 5: deterministic split-K reduce + norm/temperature scale + softmax.
// One block per (bh, c) row; writes attn padded to 64 (pad = 0) as bf16.
// ---------------------------------------------------------------------------
__global__ __launch_bounds__(64)
void k_softmax(const float* __restrict__ Sp, const float* __restrict__ ninv,
               const float* __restrict__ temp, bf16_t* __restrict__ attn) {
  __shared__ float red[64];
  int bh = blockIdx.x / CH_, c = blockIdx.x % CH_;
  int b = bh >> 2, hd = bh & 3;
  int d = threadIdx.x;
  float logit = -3.0e38f;
  if (d < CH_) {
    float s = 0.f;
    for (int sp = 0; sp < KSPLIT; ++sp)
      s += Sp[(((size_t)bh * KSPLIT + sp) * CH_ + c) * CH_ + d];
    logit = s * temp[hd] * ninv[b * C_ + hd * CH_ + c] *
            ninv[B_ * C_ + b * C_ + hd * CH_ + d];
  }
  red[d] = logit;
  __syncthreads();
  for (int st = 32; st > 0; st >>= 1) {
    if (d < st) red[d] = fmaxf(red[d], red[d + st]);
    __syncthreads();
  }
  float mx = red[0];
  __syncthreads();
  float e = (d < CH_) ? __expf(logit - mx) : 0.f;
  red[d] = e;
  __syncthreads();
  for (int st = 32; st > 0; st >>= 1) {
    if (d < st) red[d] += red[d + st];
    __syncthreads();
  }
  float inv = 1.f / red[0];
  attn[(size_t)(bh * CH_ + c) * 64 + d] =
      (d < CH_) ? (bf16_t)(e * inv) : (bf16_t)0.f;
}

// ---------------------------------------------------------------------------
// Stage 6: v channel-major -> vT[bh][pixel][64] bf16 (channels padded to 64)
// ---------------------------------------------------------------------------
__global__ __launch_bounds__(256)
void k_vt(const bf16_t* __restrict__ dwb, bf16_t* __restrict__ vT) {
  int bh = blockIdx.y;
  int b = bh >> 2, hd = bh & 3;
  int p = blockIdx.x * 256 + threadIdx.x;
  const bf16_t* src =
      dwb + (size_t)(2 * B_ * C_ + b * C_ + hd * CH_) * N_ + p;
  v8bf out8[8];
#pragma unroll
  for (int ch = 0; ch < 64; ++ch) {
    bf16_t v = (ch < CH_) ? src[(size_t)ch * N_] : (bf16_t)0.f;
    out8[ch >> 3][ch & 7] = v;
  }
  bf16_t* dst = vT + ((size_t)bh * N_ + p) * 64;
#pragma unroll
  for (int i = 0; i < 8; ++i) *(v8bf*)(dst + i * 8) = out8[i];
}

// ---------------------------------------------------------------------------
// Stage 7: out(48 x 16384) = attn(48x64) @ vT^T per (b,head)   [K=64, 2 steps]
// Block 192 thr = 6 waves: 3 M-tiles x 2 N-subtiles.
// ---------------------------------------------------------------------------
__global__ __launch_bounds__(192)
void k_out(const bf16_t* __restrict__ attn, const bf16_t* __restrict__ vT,
           float* __restrict__ out) {
  int tid = threadIdx.x, lane = tid & 31, wave = tid >> 5;  // 0..5
  int sel = lane >> 4, lm = lane & 15;
  int mt = wave % 3, ns = wave / 3;
  int bh = blockIdx.y;
  int b = bh >> 2, hd = bh & 3;
  int n0 = blockIdx.x * 32 + ns * 16;
  const bf16_t* arow = attn + (size_t)(bh * CH_ + mt * 16 + lm) * 64;
  const bf16_t* bcol = vT + ((size_t)bh * N_ + n0 + lm) * 64;
  v8f acc = vzero8f();
#pragma unroll
  for (int kk = 0; kk < 2; ++kk) {
    int k0 = kk * 32;
    V16U a;
    a.h[0] = *(const v8bf*)(arow + k0 + sel * 16);
    a.h[1] = *(const v8bf*)(arow + k0 + sel * 16 + 8);
    V16U bb;
    bb.v = *(const v16bf*)(bcol + k0 + sel * 16);
    acc = wmma_bf16(a.v, bb.v, acc);
  }
#pragma unroll
  for (int g = 0; g < 8; ++g) {
    int ch = hd * CH_ + mt * 16 + g + sel * 8;
    out[((size_t)b * C_ + ch) * N_ + n0 + lm] = acc[g];
  }
}

// ---------------------------------------------------------------------------
// Workspace layout (bytes). Total required: 176,381,952 (~168 MB).
//   [0, 75.5M)      qkv_raw bf16   -> reused after dwconv for vT (33.5M)
//   [75.5M, 151M)   qkv_dw bf16
//   [151M, +216K)   Wc bf16
//   [.., +25.2M)    xT bf16        -> reused after gemm1 for Sp/attn/ninv
// ---------------------------------------------------------------------------
#define SZ_RAW  (3ull * B_ * C_ * N_ * 2)          // 75497472
#define OFF_DW  SZ_RAW
#define OFF_WC  (OFF_DW + SZ_RAW)                  // 150994944
#define SZ_WC   (576ull * 192 * 2)                 // 221184
#define OFF_XT  (OFF_WC + SZ_WC)                   // 151216128 (32B aligned)
#define SZ_SP   (16ull * KSPLIT * CH_ * CH_ * 4)   // 9437184
#define SZ_ATTN (16ull * CH_ * 64 * 2)             // 98304

extern "C" void kernel_launch(void* const* d_in, const int* in_sizes, int n_in,
                              void* d_out, int out_size, void* d_ws,
                              size_t ws_size, hipStream_t stream) {
  (void)in_sizes; (void)n_in; (void)out_size; (void)ws_size;
  const float* x      = (const float*)d_in[0];
  const float* wq     = (const float*)d_in[1];
  const float* wq_dw  = (const float*)d_in[2];
  const float* wk     = (const float*)d_in[3];
  const float* wk_dw  = (const float*)d_in[4];
  const float* wv     = (const float*)d_in[5];
  const float* wv_dw  = (const float*)d_in[6];
  const float* temp   = (const float*)d_in[7];
  float* out = (float*)d_out;

  char* ws = (char*)d_ws;
  bf16_t* raw  = (bf16_t*)(ws);                      // stage-1 out
  bf16_t* vT   = (bf16_t*)(ws);                      // reuse after dwconv
  bf16_t* dwb  = (bf16_t*)(ws + OFF_DW);
  bf16_t* Wc   = (bf16_t*)(ws + OFF_WC);
  bf16_t* xT   = (bf16_t*)(ws + OFF_XT);
  float*  Sp   = (float*)(ws + OFF_XT);              // reuse after gemm1
  bf16_t* attn = (bf16_t*)(ws + OFF_XT + SZ_SP);
  float*  ninv = (float*)(ws + OFF_XT + SZ_SP + SZ_ATTN);

  k_xt     <<<NPIX / 256, 256, 0, stream>>>(x, xT);
  k_wc     <<<(3 * C_ * C_) / 256, 256, 0, stream>>>(wq, wk, wv, Wc);
  k_gemm1  <<<dim3(NPIX / 128, 576 / 64), 256, 0, stream>>>(Wc, xT, raw);
  k_dw     <<<3 * B_ * C_ * 64, 256, 0, stream>>>(raw, wq_dw, wk_dw, wv_dw, dwb);
  k_norm   <<<2 * B_ * C_, 256, 0, stream>>>(dwb, ninv);
  k_gram   <<<dim3(KSPLIT, B_ * HD_), 288, 0, stream>>>(dwb, Sp);
  k_softmax<<<B_ * HD_ * CH_, 64, 0, stream>>>(Sp, ninv, temp, attn);
  k_vt     <<<dim3(N_ / 256, B_ * HD_), 256, 0, stream>>>(dwb, vT);
  k_out    <<<dim3(N_ / 32, B_ * HD_), 192, 0, stream>>>(attn, vT, out);
}
